// Decoder_5033701671689
// MI455X (gfx1250) — matrix operand order repaired and checked by reading, not compiled, not run on hardware
//
#include <hip/hip_runtime.h>
#include <hip/hip_bf16.h>

typedef __bf16 bf16;
typedef __bf16 bf16x16 __attribute__((ext_vector_type(16)));
typedef float  f32x8   __attribute__((ext_vector_type(8)));

#define WMMA_BF16(a,b,c) \
  __builtin_amdgcn_wmma_f32_16x16x32_bf16(false,(a),false,(b),(short)0,(c),false,false)

#define BATCH 64
#define GROUPS 4
#define GB 16          // batch rows per group (WMMA M)
#define HID 512
#define DIMS 512
#define XDIM 1536      // DIMS + 2*HID
#define GATES 2048     // 4*HID
#define SRC 500
#define SPAD 512
#define TGT 500
#define VOCAB 3

union Frag { bf16x16 v; uint4 u[2]; };

// A matrix fragment (16x32 bf16), row-major source, contiguous K.
static __device__ __forceinline__ bf16x16 load_a(const bf16* __restrict__ p, int ld,
                                                 int r0, int k0, int lane) {
  int row = r0 + (lane & 15);
  int k   = k0 + ((lane & 16) ? 16 : 0);
  const bf16* q = p + (size_t)row * ld + k;
  Frag f;
  f.u[0] = *(const uint4*)(q);
  f.u[1] = *(const uint4*)(q + 8);
  return f.v;
}
// A fragment where every row m equals vec[k] (broadcast row vector).
static __device__ __forceinline__ bf16x16 load_a_bcast(const bf16* __restrict__ vec,
                                                       int k0, int lane) {
  int k = k0 + ((lane & 16) ? 16 : 0);
  Frag f;
  f.u[0] = *(const uint4*)(vec + k);
  f.u[1] = *(const uint4*)(vec + k + 8);
  return f.v;
}
// B matrix fragment (32x16 bf16): B[k][n] = M[n][k], M row-major [N][K].
static __device__ __forceinline__ bf16x16 load_b(const bf16* __restrict__ p, int ld,
                                                 int n0, int k0, int lane) {
  int col = n0 + (lane & 15);
  int k   = k0 + ((lane & 16) ? 16 : 0);
  const uint4* q = (const uint4*)(p + (size_t)col * ld + k);
  Frag f;
  f.u[0] = q[0];
  f.u[1] = q[1];
  return f.v;
}
// B fragment where every column n equals vec[k] (broadcast column vector).
static __device__ __forceinline__ bf16x16 load_b_bcast(const bf16* __restrict__ vec,
                                                       int k0, int lane) {
  int k = k0 + ((lane & 16) ? 16 : 0);
  const uint4* q = (const uint4*)(vec + k);
  Frag f;
  f.u[0] = q[0];
  f.u[1] = q[1];
  return f.v;
}

static __device__ __forceinline__ float sigmoidf_(float x) {
  return 1.0f / (1.0f + __expf(-x));
}

// ---------------- one-time prep kernels ----------------

__global__ void k_cvt_bf16(const float* __restrict__ s, bf16* __restrict__ d, int n) {
  int i = blockIdx.x * 256 + threadIdx.x;
  if (i < n) d[i] = (bf16)s[i];
}
__global__ void k_copy_f32(const float* __restrict__ s, float* __restrict__ d, int n) {
  int i = blockIdx.x * 256 + threadIdx.x;
  if (i < n) d[i] = s[i];
}
__global__ void k_copy_i32(const int* __restrict__ s, int* __restrict__ d, int n) {
  int i = blockIdx.x * 256 + threadIdx.x;
  if (i < n) d[i] = s[i];
}
__global__ void k_add2(const float* __restrict__ a, const float* __restrict__ b,
                       float* __restrict__ d, int n) {
  int i = blockIdx.x * 256 + threadIdx.x;
  if (i < n) d[i] = a[i] + b[i];
}
// ctx2[b][s][d] = ctx[b][s][d] + ctx[b][s][d+512]  (folded duplicated query), s>=500 -> 0
__global__ void k_ctx2(const float* __restrict__ ctx, bf16* __restrict__ dst) {
  int i = blockIdx.x * 256 + threadIdx.x;        // 64*512*512
  int d = i & 511, s = (i >> 9) & 511, b = i >> 18;
  float v = 0.0f;
  if (s < SRC) {
    const float* r = ctx + ((size_t)b * SRC + s) * 1024;
    v = r[d] + r[d + 512];
  }
  dst[i] = (bf16)v;
}
// ctxT[b][d][s] = ctx[b][s][d], s>=500 -> 0
__global__ void k_ctxT(const float* __restrict__ ctx, bf16* __restrict__ dst) {
  int i = blockIdx.x * 256 + threadIdx.x;        // 64*1024*512
  int s = i & 511, d = (i >> 9) & 1023, b = i >> 19;
  float v = 0.0f;
  if (s < SRC) v = ctx[((size_t)b * SRC + s) * 1024 + d];
  dst[i] = (bf16)v;
}

// ---------------- persistent decoder kernel ----------------

__global__ __launch_bounds__(1024, 1) void decoder_main(
    const bf16* __restrict__ ctx2,   // [64][512][512]
    const bf16* __restrict__ ctxT,   // [64][1024][512]
    const bf16* __restrict__ Wih0,   // [2048][1536]
    const bf16* __restrict__ Whh0,   // [2048][512]
    const bf16* __restrict__ Wih1,   // [2048][512]
    const bf16* __restrict__ Whh1,   // [2048][512]
    const bf16* __restrict__ embb,   // [3][512]
    const float* __restrict__ b0,    // [2048]
    const float* __restrict__ b1,    // [2048]
    const float* __restrict__ Wl,    // [3][512] fp32
    const float* __restrict__ bl,    // [3]
    const int*   __restrict__ target,// [64][500]
    float* __restrict__ c1,          // [64][512]
    float* __restrict__ c2,          // [64][512]
    bf16*  __restrict__ h1b,         // [64][512]
    bf16*  __restrict__ h2b,         // [64][512]
    bf16*  __restrict__ xbf,         // [64][1536]
    float* __restrict__ scores,      // [64][512]
    bf16*  __restrict__ wbf,         // [64][512]
    float* __restrict__ out)         // [64][500][3]
{
  const int g   = blockIdx.x;       // batch group 0..3
  const int bg0 = g * GB;           // first global batch row
  const int tid = threadIdx.x;
  const int wave = tid >> 5;
  const int lane = tid & 31;

  __shared__ int   nextidx[GB];
  __shared__ float slog[GB][VOCAB];

  if (tid < GB) nextidx[tid] = target[(size_t)(bg0 + tid) * TGT];
  __syncthreads();

#pragma unroll 1
  for (int t = 0; t < TGT; ++t) {
    // ---- S0: embedding -> xbf[:, 0:512] (bf16) ----
    for (int e = tid; e < GB * DIMS / 2; e += 1024) {  // 4096 dword copies
      int b = e >> 8, c = e & 255;
      const unsigned int* src = (const unsigned int*)(embb + (size_t)nextidx[b] * DIMS);
      ((unsigned int*)(xbf + (size_t)(bg0 + b) * XDIM))[c] = src[c];
    }

    // ---- S1: attention scores: scores[b][s] = sum_d ctx2[b][s][d]*h2[b][d] ----
    for (int task = wave; task < GB * (SPAD / 16); task += 32) {
      int b = task >> 5, st = task & 31;
      f32x8 acc = {0.f, 0.f, 0.f, 0.f, 0.f, 0.f, 0.f, 0.f};
      const bf16* arow = ctx2 + (size_t)(bg0 + b) * SPAD * 512;
      const bf16* hvec = h2b + (size_t)(bg0 + b) * HID;
      for (int kt = 0; kt < 512; kt += 32) {
        bf16x16 a  = load_a(arow, 512, st * 16, kt, lane);
        bf16x16 bb = load_b_bcast(hvec, kt, lane);
        acc = WMMA_BF16(a, bb, acc);
      }
      if ((lane & 15) == 0) {
        float* sd = scores + (size_t)(bg0 + b) * SPAD + st * 16 + ((lane >> 4) * 8);
#pragma unroll
        for (int i = 0; i < 8; ++i) sd[i] = acc[i];
      }
    }
    __threadfence(); __syncthreads();

    // ---- S2: softmax over s (wave b handles batch row b) ----
    if (wave < GB) {
      int b = wave;
      const float* sp = scores + (size_t)(bg0 + b) * SPAD;
      float vals[16];
      float mx = -1e30f;
#pragma unroll
      for (int j = 0; j < 16; ++j) {
        int s = j * 32 + lane;
        float v = (s < SRC) ? sp[s] : -1e30f;
        vals[j] = v;
        mx = fmaxf(mx, v);
      }
      for (int o = 16; o > 0; o >>= 1) mx = fmaxf(mx, __shfl_xor(mx, o, 32));
      float sum = 0.f;
#pragma unroll
      for (int j = 0; j < 16; ++j) { float e = __expf(vals[j] - mx); vals[j] = e; sum += e; }
      for (int o = 16; o > 0; o >>= 1) sum += __shfl_xor(sum, o, 32);
      float inv = 1.0f / sum;
      bf16* wp = wbf + (size_t)(bg0 + b) * SPAD;
#pragma unroll
      for (int j = 0; j < 16; ++j) {
        int s = j * 32 + lane;
        wp[s] = (bf16)((s < SRC) ? vals[j] * inv : 0.0f);
      }
    }
    __threadfence(); __syncthreads();

    // ---- S3: align[b][d] = sum_s w[b][s]*ctx[b][s][d]  -> xbf[:, 512:1536] ----
    for (int task = wave; task < GB * 64; task += 32) {
      int b = task >> 6, dt = task & 63;
      f32x8 acc = {0.f, 0.f, 0.f, 0.f, 0.f, 0.f, 0.f, 0.f};
      const bf16* wv = wbf + (size_t)(bg0 + b) * SPAD;
      const bf16* ct = ctxT + (size_t)(bg0 + b) * 1024 * 512;
      for (int kt = 0; kt < 512; kt += 32) {
        bf16x16 a  = load_a_bcast(wv, kt, lane);
        bf16x16 bb = load_b(ct, 512, dt * 16, kt, lane);
        acc = WMMA_BF16(a, bb, acc);
      }
      if (lane < 16)
        xbf[(size_t)(bg0 + b) * XDIM + DIMS + dt * 16 + lane] = (bf16)acc[0];
    }
    __threadfence(); __syncthreads();

    // ---- S4: LSTM cell 0: gates = x@Wih0^T + h1@Whh0^T + b0 ----
    {
      int jt = wave;  // hidden-unit tile 0..31
      f32x8 ai = {0,0,0,0,0,0,0,0}, af = {0,0,0,0,0,0,0,0};
      f32x8 ag = {0,0,0,0,0,0,0,0}, ao = {0,0,0,0,0,0,0,0};
      const bf16* xg = xbf + (size_t)bg0 * XDIM;
      for (int kt = 0; kt < XDIM; kt += 32) {
        bf16x16 a = load_a(xg, XDIM, 0, kt, lane);
        ai = WMMA_BF16(a, load_b(Wih0, XDIM,        jt * 16, kt, lane), ai);
        af = WMMA_BF16(a, load_b(Wih0, XDIM,  512 + jt * 16, kt, lane), af);
        ag = WMMA_BF16(a, load_b(Wih0, XDIM, 1024 + jt * 16, kt, lane), ag);
        ao = WMMA_BF16(a, load_b(Wih0, XDIM, 1536 + jt * 16, kt, lane), ao);
      }
      const bf16* hg = h1b + (size_t)bg0 * HID;
      for (int kt = 0; kt < HID; kt += 32) {
        bf16x16 a = load_a(hg, HID, 0, kt, lane);
        ai = WMMA_BF16(a, load_b(Whh0, HID,        jt * 16, kt, lane), ai);
        af = WMMA_BF16(a, load_b(Whh0, HID,  512 + jt * 16, kt, lane), af);
        ag = WMMA_BF16(a, load_b(Whh0, HID, 1024 + jt * 16, kt, lane), ag);
        ao = WMMA_BF16(a, load_b(Whh0, HID, 1536 + jt * 16, kt, lane), ao);
      }
      __syncthreads();  // all reads of h1b done before updates
      int n = lane & 15, j = jt * 16 + n, mb = (lane & 16) ? 8 : 0;
#pragma unroll
      for (int i = 0; i < 8; ++i) {
        int br = bg0 + mb + i;
        float I = sigmoidf_(ai[i] + b0[j]);
        float F = sigmoidf_(af[i] + b0[512 + j]);
        float G = tanhf(ag[i] + b0[1024 + j]);
        float O = sigmoidf_(ao[i] + b0[1536 + j]);
        float c  = c1[(size_t)br * HID + j];
        float cn = F * c + I * G;
        c1[(size_t)br * HID + j] = cn;
        h1b[(size_t)br * HID + j] = (bf16)(O * tanhf(cn));
      }
    }
    __threadfence(); __syncthreads();

    // ---- S5: LSTM cell 1: gates = h1n@Wih1^T + h2@Whh1^T + b1 ----
    {
      int jt = wave;
      f32x8 ai = {0,0,0,0,0,0,0,0}, af = {0,0,0,0,0,0,0,0};
      f32x8 ag = {0,0,0,0,0,0,0,0}, ao = {0,0,0,0,0,0,0,0};
      const bf16* h1g = h1b + (size_t)bg0 * HID;
      for (int kt = 0; kt < HID; kt += 32) {
        bf16x16 a = load_a(h1g, HID, 0, kt, lane);
        ai = WMMA_BF16(a, load_b(Wih1, HID,        jt * 16, kt, lane), ai);
        af = WMMA_BF16(a, load_b(Wih1, HID,  512 + jt * 16, kt, lane), af);
        ag = WMMA_BF16(a, load_b(Wih1, HID, 1024 + jt * 16, kt, lane), ag);
        ao = WMMA_BF16(a, load_b(Wih1, HID, 1536 + jt * 16, kt, lane), ao);
      }
      const bf16* h2g = h2b + (size_t)bg0 * HID;
      for (int kt = 0; kt < HID; kt += 32) {
        bf16x16 a = load_a(h2g, HID, 0, kt, lane);
        ai = WMMA_BF16(a, load_b(Whh1, HID,        jt * 16, kt, lane), ai);
        af = WMMA_BF16(a, load_b(Whh1, HID,  512 + jt * 16, kt, lane), af);
        ag = WMMA_BF16(a, load_b(Whh1, HID, 1024 + jt * 16, kt, lane), ag);
        ao = WMMA_BF16(a, load_b(Whh1, HID, 1536 + jt * 16, kt, lane), ao);
      }
      __syncthreads();  // all reads of h2b done before updates
      int n = lane & 15, j = jt * 16 + n, mb = (lane & 16) ? 8 : 0;
#pragma unroll
      for (int i = 0; i < 8; ++i) {
        int br = bg0 + mb + i;
        float I = sigmoidf_(ai[i] + b1[j]);
        float F = sigmoidf_(af[i] + b1[512 + j]);
        float G = tanhf(ag[i] + b1[1024 + j]);
        float O = sigmoidf_(ao[i] + b1[1536 + j]);
        float c  = c2[(size_t)br * HID + j];
        float cn = F * c + I * G;
        c2[(size_t)br * HID + j] = cn;
        h2b[(size_t)br * HID + j] = (bf16)(O * tanhf(cn));
      }
    }
    __threadfence(); __syncthreads();

    // ---- S6: logits, log_softmax output, argmax -> next input ----
    if (tid < GB * VOCAB) {
      int b = tid / VOCAB, v = tid - b * VOCAB;
      const bf16*  h = h2b + (size_t)(bg0 + b) * HID;
      const float* w = Wl + (size_t)v * HID;
      float s = 0.f;
      for (int jj = 0; jj < HID; ++jj) s += (float)h[jj] * w[jj];
      slog[b][v] = s + bl[v];
    }
    __syncthreads();
    if (tid < GB) {
      int b = tid;
      float l0 = slog[b][0], l1 = slog[b][1], l2 = slog[b][2];
      int am = 0; float bm = l0;
      if (l1 > bm) { bm = l1; am = 1; }
      if (l2 > bm) { bm = l2; am = 2; }
      nextidx[b] = am;
      float lse = bm + __logf(__expf(l0 - bm) + __expf(l1 - bm) + __expf(l2 - bm));
      size_t o = ((size_t)(bg0 + b) * TGT + t) * VOCAB;
      out[o + 0] = l0 - lse;
      out[o + 1] = l1 - lse;
      out[o + 2] = l2 - lse;
    }
    __syncthreads();
  }
}

// ---------------- host launcher ----------------

extern "C" void kernel_launch(void* const* d_in, const int* in_sizes, int n_in,
                              void* d_out, int out_size, void* d_ws, size_t ws_size,
                              hipStream_t stream) {
  const float* ctx    = (const float*)d_in[0];
  const int*   target = (const int*)d_in[1];
  const float* h1_0   = (const float*)d_in[3];
  const float* c1_0   = (const float*)d_in[4];
  const float* h2_0   = (const float*)d_in[5];
  const float* c2_0   = (const float*)d_in[6];
  const float* emb    = (const float*)d_in[7];
  const float* W_ih0  = (const float*)d_in[8];
  const float* W_hh0  = (const float*)d_in[9];
  const float* b_ih0  = (const float*)d_in[10];
  const float* b_hh0  = (const float*)d_in[11];
  const float* W_ih1  = (const float*)d_in[12];
  const float* W_hh1  = (const float*)d_in[13];
  const float* b_ih1  = (const float*)d_in[14];
  const float* b_hh1  = (const float*)d_in[15];
  const float* Wl     = (const float*)d_in[16];
  const float* bl     = (const float*)d_in[17];

  char* p = (char*)d_ws;
  auto alloc = [&](size_t bytes) -> void* {
    void* r = (void*)p;
    p += (bytes + 255) & ~(size_t)255;
    return r;
  };
  bf16*  ctx2  = (bf16*)alloc((size_t)64 * 512 * 512 * 2);
  bf16*  ctxT  = (bf16*)alloc((size_t)64 * 1024 * 512 * 2);
  bf16*  Wih0b = (bf16*)alloc((size_t)GATES * XDIM * 2);
  bf16*  Whh0b = (bf16*)alloc((size_t)GATES * HID * 2);
  bf16*  Wih1b = (bf16*)alloc((size_t)GATES * HID * 2);
  bf16*  Whh1b = (bf16*)alloc((size_t)GATES * HID * 2);
  bf16*  embb  = (bf16*)alloc((size_t)VOCAB * DIMS * 2);
  float* b0    = (float*)alloc((size_t)GATES * 4);
  float* b1    = (float*)alloc((size_t)GATES * 4);
  float* c1    = (float*)alloc((size_t)BATCH * HID * 4);
  float* c2    = (float*)alloc((size_t)BATCH * HID * 4);
  bf16*  h1b   = (bf16*)alloc((size_t)BATCH * HID * 2);
  bf16*  h2b   = (bf16*)alloc((size_t)BATCH * HID * 2);
  bf16*  xbf   = (bf16*)alloc((size_t)BATCH * XDIM * 2);
  float* sc    = (float*)alloc((size_t)BATCH * SPAD * 4);
  bf16*  wbf   = (bf16*)alloc((size_t)BATCH * SPAD * 2);
  if ((size_t)(p - (char*)d_ws) > ws_size) return;  // workspace too small

  // one-time prep (graph-captured; trivial vs the recurrence)
  k_ctx2<<<(64 * 512 * 512) / 256, 256, 0, stream>>>(ctx, ctx2);
  k_ctxT<<<(64 * 1024 * 512) / 256, 256, 0, stream>>>(ctx, ctxT);
  k_cvt_bf16<<<(GATES * XDIM + 255) / 256, 256, 0, stream>>>(W_ih0, Wih0b, GATES * XDIM);
  k_cvt_bf16<<<(GATES * HID + 255) / 256, 256, 0, stream>>>(W_hh0, Whh0b, GATES * HID);
  k_cvt_bf16<<<(GATES * HID + 255) / 256, 256, 0, stream>>>(W_ih1, Wih1b, GATES * HID);
  k_cvt_bf16<<<(GATES * HID + 255) / 256, 256, 0, stream>>>(W_hh1, Whh1b, GATES * HID);
  k_cvt_bf16<<<(VOCAB * DIMS + 255) / 256, 256, 0, stream>>>(emb, embb, VOCAB * DIMS);
  k_add2<<<(GATES + 255) / 256, 256, 0, stream>>>(b_ih0, b_hh0, b0, GATES);
  k_add2<<<(GATES + 255) / 256, 256, 0, stream>>>(b_ih1, b_hh1, b1, GATES);
  k_cvt_bf16<<<(BATCH * HID + 255) / 256, 256, 0, stream>>>(h1_0, h1b, BATCH * HID);
  k_cvt_bf16<<<(BATCH * HID + 255) / 256, 256, 0, stream>>>(h2_0, h2b, BATCH * HID);
  k_copy_f32<<<(BATCH * HID + 255) / 256, 256, 0, stream>>>(c1_0, c1, BATCH * HID);
  k_copy_f32<<<(BATCH * HID + 255) / 256, 256, 0, stream>>>(c2_0, c2, BATCH * HID);

  // second tuple output: target_ passthrough (after the [64][500][3] logits)
  if (out_size >= BATCH * TGT * VOCAB + BATCH * TGT) {
    k_copy_i32<<<(BATCH * TGT + 255) / 256, 256, 0, stream>>>(
        target, ((int*)d_out) + BATCH * TGT * VOCAB, BATCH * TGT);
  }

  decoder_main<<<GROUPS, 1024, 0, stream>>>(
      ctx2, ctxT, Wih0b, Whh0b, Wih1b, Whh1b, embb, b0, b1, Wl, bl, target,
      c1, c2, h1b, h2b, xbf, sc, wbf, (float*)d_out);
}
